// TransformerBlock_6176162972318
// MI455X (gfx1250) — matrix operand order repaired and checked by reading, not compiled, not run on hardware
//
#include <hip/hip_runtime.h>
#include <cstdint>

#define DEV __device__ __forceinline__

typedef unsigned short u16;
typedef __attribute__((ext_vector_type(16))) __bf16 v16bf;
typedef __attribute__((ext_vector_type(8)))  float  v8f;
typedef uint32_t su32x4 __attribute__((ext_vector_type(4)));
typedef uint32_t su32x8 __attribute__((ext_vector_type(8)));

union Frag {
    v16bf v;
    uint4 q[2];
    u16   u[16];
};

DEV u16 f2bf(float f) {
    uint32_t x = __builtin_bit_cast(uint32_t, f);
    uint32_t r = (x + 0x7FFFu + ((x >> 16) & 1u)) >> 16;
    return (u16)r;
}

DEV v8f wmma_bf16(const Frag& a, const Frag& b, v8f c) {
    // D(16x16 f32) = A(16x32 bf16) * B(32x16 bf16) + C
    return __builtin_amdgcn_wmma_f32_16x16x32_bf16(
        false, a.v, false, b.v, (short)0, c, false, false);
}

// ---------------------------------------------------------------------------
// TDM: issue a 2D tensor_load_to_lds (ISA ch.8: D# groups 0/1, 2-group form).
// s0flags carries data_size + pad_enable/interval/amount (LDS row padding).
// ---------------------------------------------------------------------------
DEV void tdm_load_2d(const u16* gptr, uint32_t lds_off,
                     uint32_t dim0, uint32_t dim1,
                     uint32_t tile0, uint32_t tile1,
                     uint32_t stride0, uint32_t s0flags) {
    const uint64_t ga = (uint64_t)(uintptr_t)gptr;
    su32x4 g0;
    g0[0] = 1u;                                   // count=1, user descriptor
    g0[1] = lds_off;                              // lds_addr [63:32]
    g0[2] = (uint32_t)ga;                         // global_addr lo
    g0[3] = (uint32_t)(ga >> 32) | 0x80000000u;   // global_addr hi | type=2
    su32x8 g1;
    g1[0] = s0flags;                              // mask=0,data_size=2B,pad cfg
    g1[1] = (dim0 & 0xFFFFu) << 16;               // tensor_dim0[15:0] @bit48
    g1[2] = (dim0 >> 16) | ((dim1 & 0xFFFFu) << 16);
    g1[3] = (dim1 >> 16) | (tile0 << 16);         // tile_dim0 @bit112
    g1[4] = tile1;                                // tile_dim1
    g1[5] = stride0;                              // tensor_dim0_stride[31:0]
    g1[6] = 0u;
    g1[7] = 0u;
    asm volatile("tensor_load_to_lds %0, %1" :: "s"(g0), "s"(g1) : "memory");
}

// pad: data_size=2B | pad_enable | interval | amount(4 DW = 16B per row)
#define TDM_FLAGS_A 0x06D10000u  // interval code 3 (16 DW = 64B row), +16B pad
#define TDM_FLAGS_B 0x07110000u  // interval code 4 (32 DW = 128B row), +16B pad

// ---------------------------------------------------------------------------
// fp32 -> bf16 conversion
// ---------------------------------------------------------------------------
__global__ __launch_bounds__(256) void cvt_bf16_kernel(const float* __restrict__ in,
                                                       u16* __restrict__ out, int n) {
    int i = blockIdx.x * 256 + threadIdx.x;
    if (i < n) out[i] = f2bf(in[i]);
}

// ---------------------------------------------------------------------------
// LayerNorm over D=1024; writes bf16 + optional fp32
// ---------------------------------------------------------------------------
__global__ __launch_bounds__(256) void ln_kernel(const float* __restrict__ x,
                                                 const float* __restrict__ g,
                                                 const float* __restrict__ be,
                                                 u16* __restrict__ outB,
                                                 float* __restrict__ outF) {
    __shared__ float red[256];
    const int tid = threadIdx.x;
    const size_t base = (size_t)blockIdx.x << 10;
    float v[4];
    float s = 0.f;
#pragma unroll
    for (int i = 0; i < 4; ++i) { v[i] = x[base + tid + (i << 8)]; s += v[i]; }
    red[tid] = s; __syncthreads();
    for (int off = 128; off > 0; off >>= 1) {
        if (tid < off) red[tid] += red[tid + off];
        __syncthreads();
    }
    const float mean = red[0] * (1.f / 1024.f);
    __syncthreads();
    s = 0.f;
#pragma unroll
    for (int i = 0; i < 4; ++i) { float d = v[i] - mean; s += d * d; }
    red[tid] = s; __syncthreads();
    for (int off = 128; off > 0; off >>= 1) {
        if (tid < off) red[tid] += red[tid + off];
        __syncthreads();
    }
    const float rstd = rsqrtf(red[0] * (1.f / 1024.f) + 1e-5f);
#pragma unroll
    for (int i = 0; i < 4; ++i) {
        const int c = tid + (i << 8);
        const float nv = (v[i] - mean) * rstd * g[c] + be[c];
        outB[base + c] = f2bf(nv);
        if (outF) outF[base + c] = nv;
    }
}

// ---------------------------------------------------------------------------
// WMMA GEMM, TDM-fed + double buffered.
// C[M,N] = A_bf16[M,K] @ W_bf16[K,N] + bias, templated epilogue.
// 64x64 tile / WG, 4 waves; K-steps of 32. Wave0 issues tensor_load_to_lds for
// tile k+1, s_wait_tensorcnt(2) retires tile k, barrier, WMMA on tile k.
// LDS rows padded by TDM (+16B) => A pitch 40 u16, B pitch 72 u16.
// ---------------------------------------------------------------------------
enum { EPI_BF16 = 0, EPI_KTRANS = 1, EPI_F32_RES = 2, EPI_GELU = 3, EPI_F32_RES2 = 5 };

template <int EPI>
__global__ __launch_bounds__(128) void gemm_wmma(
    const u16* __restrict__ A, const u16* __restrict__ W,
    const float* __restrict__ bias,
    const float* __restrict__ res1, const float* __restrict__ res2,
    float* __restrict__ outF, u16* __restrict__ outB,
    int M, int N, int K) {
    constexpr int AS = 40;   // padded LDS pitch (u16) for 32-elem A rows
    constexpr int BS = 72;   // padded LDS pitch (u16) for 64-elem B rows
    __shared__ u16 sA[2][64 * AS];
    __shared__ u16 sB[2][32 * BS];

    const int tid  = threadIdx.x;
    const int wave = tid >> 5, lane = tid & 31;
    const int row0 = blockIdx.y * 64, col0 = blockIdx.x * 64;

    v8f acc[4] = {};
    const int k0   = (lane >> 4) << 3;          // A-fragment K-chunk base
    const int mloc = (wave << 4) + (lane & 15); // A-fragment row within tile
    const int KT   = K >> 5;

    const uint32_t ldsA0 = (uint32_t)(uintptr_t)&sA[0][0];
    const uint32_t ldsA1 = (uint32_t)(uintptr_t)&sA[1][0];
    const uint32_t ldsB0 = (uint32_t)(uintptr_t)&sB[0][0];
    const uint32_t ldsB1 = (uint32_t)(uintptr_t)&sB[1][0];

    if (wave == 0) {  // prologue: fill buffer 0 with tile 0
        tdm_load_2d(A + (size_t)row0 * K, ldsA0, (uint32_t)K, (uint32_t)M,
                    32u, 64u, (uint32_t)K, TDM_FLAGS_A);
        tdm_load_2d(W + col0, ldsB0, (uint32_t)N, (uint32_t)K,
                    64u, 32u, (uint32_t)N, TDM_FLAGS_B);
    }

    for (int kt = 0; kt < KT; ++kt) {
        const int buf = kt & 1;
        if (wave == 0) {
            if (kt + 1 < KT) {  // overlap: DMA tile kt+1 into the other buffer
                const int kb = (kt + 1) << 5;
                tdm_load_2d(A + (size_t)row0 * K + kb, buf ? ldsA0 : ldsA1,
                            (uint32_t)K, (uint32_t)M, 32u, 64u, (uint32_t)K, TDM_FLAGS_A);
                tdm_load_2d(W + (size_t)kb * N + col0, buf ? ldsB0 : ldsB1,
                            (uint32_t)N, (uint32_t)K, 64u, 32u, (uint32_t)N, TDM_FLAGS_B);
                __builtin_amdgcn_s_wait_tensorcnt(2);  // tile kt retired
            } else {
                __builtin_amdgcn_s_wait_tensorcnt(0);
            }
        }
        __syncthreads();        // tile kt visible to all waves

        Frag fa;
        fa.q[0] = *(const uint4*)&sA[buf][mloc * AS + k0];
        fa.q[1] = *(const uint4*)&sA[buf][mloc * AS + k0 + 16];
#pragma unroll
        for (int nt = 0; nt < 4; ++nt) {
            Frag fb;            // lane%16 = N column, elements over K (table order)
            for (int s = 0; s < 16; ++s)
                fb.u[s] = sB[buf][(k0 + (s & 7) + ((s >> 3) << 4)) * BS + nt * 16 + (lane & 15)];
            acc[nt] = wmma_bf16(fa, fb, acc[nt]);
        }
        __syncthreads();        // done reading buf before TDM refills it
    }

    const int rbase = (lane >> 4) << 3;
#pragma unroll
    for (int nt = 0; nt < 4; ++nt) {
#pragma unroll
        for (int v = 0; v < 8; ++v) {
            const int m = row0 + (wave << 4) + rbase + v;
            const int n = col0 + nt * 16 + (lane & 15);
            float val = acc[nt][v] + bias[n];
            const size_t o = (size_t)m * N + n;
            if constexpr (EPI == EPI_BF16) {
                outB[o] = f2bf(val);
            } else if constexpr (EPI == EPI_KTRANS) {
                // K stored [B,H,DK,S] so attention's B-fragment is contiguous
                const int bb = m >> 10, s = m & 1023;
                const int h = n >> 6, d = n & 63;
                outB[((size_t)((bb * 16 + h) * 64 + d) << 10) + s] = f2bf(val);
            } else if constexpr (EPI == EPI_F32_RES) {
                outF[o] = val + res1[o];
            } else if constexpr (EPI == EPI_GELU) {
                const float ge = 0.5f * val * (1.0f + erff(val * 0.70710678118654752f));
                outB[o] = f2bf(ge);
            } else {
                outF[o] = val + res1[o] + res2[o];
            }
        }
    }
}

// ---------------------------------------------------------------------------
// Attention: one WG (8 waves) per (b, h, 16-row Q tile).
// scores(16x1024) fp32 in LDS, softmax fp32, P bf16 in LDS, ctx = P@V.
// K/V fragment loads are software-pipelined one tile ahead of the WMMAs.
// ---------------------------------------------------------------------------
__global__ __launch_bounds__(256) void attn_kernel(
    const u16* __restrict__ Qb, const u16* __restrict__ Ktb,
    const u16* __restrict__ Vb, const float* __restrict__ rel_table,
    u16* __restrict__ ctxB) {
    __shared__ float sS[16][1024];      // scores (fp32)
    __shared__ u16   sP[16][1024];      // probs  (bf16)
    __shared__ float sPart[2][16][64];  // ctx partial sums
    __shared__ float sRed[16][16];
    __shared__ float sMax[16];
    __shared__ float sInv[16];

    const int tid  = threadIdx.x;
    const int wave = tid >> 5, lane = tid & 31;
    const int id = blockIdx.x;
    const int qt = id & 63;
    const int h  = (id >> 6) & 15;
    const int b  = id >> 10;

    const int k0    = (lane >> 4) << 3;
    const int rbase = (lane >> 4) << 3;

    // ---- phase 1: scores = Q K^T / 8 + bias --------------------------------
    Frag fq[2];
    {
        const u16* qrow = Qb + ((size_t)((b << 10) + qt * 16 + (lane & 15)) << 10) + (h << 6);
#pragma unroll
        for (int t = 0; t < 2; ++t) {   // DK=64 -> two K-steps of 32
            fq[t].q[0] = *(const uint4*)(qrow + t * 32 + k0);
            fq[t].q[1] = *(const uint4*)(qrow + t * 32 + k0 + 16);
        }
    }
    {
        const u16* kbase = Ktb + ((size_t)((b * 16 + h) * 64) << 10) + (lane & 15);
        Frag c0, c1, n0, n1;
        // prologue load for first column tile
        {
            const int j0 = wave * 128;
            for (int s = 0; s < 16; ++s) {
                const size_t kk = (size_t)(k0 + (s & 7) + ((s >> 3) << 4)) << 10;
                c0.u[s] = kbase[kk + j0];
                c1.u[s] = kbase[(32 << 10) + kk + j0];
            }
        }
#pragma unroll
        for (int i = 0; i < 8; ++i) {   // each wave: 8 column tiles of 16
            const int j0 = wave * 128 + i * 16;
            if (i < 7) {                // prefetch next tile's K fragments
                const int jn = j0 + 16;
                for (int s = 0; s < 16; ++s) {
                    const size_t kk = (size_t)(k0 + (s & 7) + ((s >> 3) << 4)) << 10;
                    n0.u[s] = kbase[kk + jn];
                    n1.u[s] = kbase[(32 << 10) + kk + jn];
                }
            }
            v8f acc = {};
            acc = wmma_bf16(fq[0], c0, acc);
            acc = wmma_bf16(fq[1], c1, acc);
#pragma unroll
            for (int v = 0; v < 8; ++v) {
                const int row = rbase + v;
                const int ig = qt * 16 + row;
                const int j  = j0 + (lane & 15);
                int rel = j - ig + 63;
                rel = rel < 0 ? 0 : (rel > 126 ? 126 : rel);
                sS[row][j] = acc[v] * 0.125f + rel_table[rel * 16 + h];
            }
            c0 = n0; c1 = n1;
        }
    }
    __syncthreads();

    // ---- phase 2: softmax (unnormalized P; 1/sum folded into ctx) ----------
    {
        const int row = tid >> 4, c = tid & 15;
        float mx = -3.0e38f;
        for (int j = c; j < 1024; j += 16) mx = fmaxf(mx, sS[row][j]);
        sRed[row][c] = mx; __syncthreads();
        if (c == 0) {
            float m = sRed[row][0];
#pragma unroll
            for (int t = 1; t < 16; ++t) m = fmaxf(m, sRed[row][t]);
            sMax[row] = m;
        }
        __syncthreads();
        const float m = sMax[row];
        float sum = 0.f;
        for (int j = c; j < 1024; j += 16) {
            const float p = __expf(sS[row][j] - m);
            sum += p;
            sP[row][j] = f2bf(p);
        }
        sRed[row][c] = sum; __syncthreads();
        if (c == 0) {
            float t0 = 0.f;
#pragma unroll
            for (int t = 0; t < 16; ++t) t0 += sRed[row][t];
            sInv[row] = 1.0f / t0;
        }
        __syncthreads();
    }

    // ---- phase 3: ctx = P @ V (V fragments pipelined) ----------------------
    {
        const int nt = wave & 3, kh = wave >> 2;
        const int d0 = nt * 16;
        const int ks0 = kh * 16;
        const u16* vcol = Vb + ((size_t)(b << 10) << 10) + (h << 6) + d0;
        Frag fb_cur, fb_nxt;
        {
            for (int s = 0; s < 16; ++s)
                fb_cur.u[s] = vcol[((size_t)(ks0 * 32 + k0 + (s & 7) + ((s >> 3) << 4)) << 10) + (lane & 15)];
        }
        v8f acc = {};
        for (int ks = ks0; ks < ks0 + 16; ++ks) {
            const int tb = ks * 32;
            if (ks + 1 < ks0 + 16) {    // prefetch next V fragment
                for (int s = 0; s < 16; ++s)
                    fb_nxt.u[s] = vcol[((size_t)(tb + 32 + k0 + (s & 7) + ((s >> 3) << 4)) << 10) + (lane & 15)];
            }
            Frag fa;
            fa.q[0] = *(const uint4*)&sP[lane & 15][tb + k0];
            fa.q[1] = *(const uint4*)&sP[lane & 15][tb + k0 + 16];
            acc = wmma_bf16(fa, fb_cur, acc);
            fb_cur = fb_nxt;
        }
#pragma unroll
        for (int v = 0; v < 8; ++v) sPart[kh][rbase + v][d0 + (lane & 15)] = acc[v];
    }
    __syncthreads();

    // ---- phase 4: reduce halves, normalize, store bf16 ---------------------
    for (int e = tid; e < 16 * 64; e += 256) {
        const int row = e >> 6, d = e & 63;
        const float val = (sPart[0][row][d] + sPart[1][row][d]) * sInv[row];
        ctxB[((size_t)((b << 10) + qt * 16 + row) << 10) + (h << 6) + d] = f2bf(val);
    }
}

// ---------------------------------------------------------------------------
// Host launcher
// ---------------------------------------------------------------------------
extern "C" void kernel_launch(void* const* d_in, const int* in_sizes, int n_in,
                              void* d_out, int out_size, void* d_ws, size_t ws_size,
                              hipStream_t stream) {
    constexpr int Bb = 4, S = 1024, D = 1024, H = 16, DFF = 4096;
    constexpr int M = Bb * S;  // 4096 rows

    const float* x   = (const float*)d_in[0];
    const float* wq  = (const float*)d_in[1];  const float* bq = (const float*)d_in[2];
    const float* wk  = (const float*)d_in[3];  const float* bk = (const float*)d_in[4];
    const float* wv  = (const float*)d_in[5];  const float* bv = (const float*)d_in[6];
    const float* wo  = (const float*)d_in[7];  const float* bo = (const float*)d_in[8];
    const float* rel = (const float*)d_in[9];
    const float* g1  = (const float*)d_in[10]; const float* be1 = (const float*)d_in[11];
    const float* g2  = (const float*)d_in[12]; const float* be2 = (const float*)d_in[13];
    const float* w1  = (const float*)d_in[14]; const float* b1  = (const float*)d_in[15];
    const float* w2  = (const float*)d_in[16]; const float* b2  = (const float*)d_in[17];
    const float* w3  = (const float*)d_in[18]; const float* b3  = (const float*)d_in[19];
    const float* gf  = (const float*)d_in[20]; const float* bfb = (const float*)d_in[21];

    char* ws = (char*)d_ws;
    size_t off = 0;
    auto alloc = [&](size_t bytes) -> void* {
        void* p = ws + off;
        off = (off + bytes + 255) & ~(size_t)255;
        return p;
    };
    u16* wq_b  = (u16*)alloc((size_t)D * D * 2);
    u16* wk_b  = (u16*)alloc((size_t)D * D * 2);
    u16* wv_b  = (u16*)alloc((size_t)D * D * 2);
    u16* wo_b  = (u16*)alloc((size_t)D * D * 2);
    u16* w1_b  = (u16*)alloc((size_t)D * DFF * 2);
    u16* w2_b  = (u16*)alloc((size_t)DFF * D * 2);
    u16* w3_b  = (u16*)alloc((size_t)D * D * 2);
    u16* nx_b  = (u16*)alloc((size_t)M * D * 2);
    u16* Q_b   = (u16*)alloc((size_t)M * D * 2);
    u16* Kt_b  = (u16*)alloc((size_t)M * D * 2);
    u16* V_b   = (u16*)alloc((size_t)M * D * 2);
    u16* ctx_b = (u16*)alloc((size_t)M * D * 2);
    u16* nx2_b = (u16*)alloc((size_t)M * D * 2);
    u16* h1_b  = (u16*)alloc((size_t)M * DFF * 2);
    u16* yn_b  = (u16*)alloc((size_t)M * D * 2);
    float* xa   = (float*)alloc((size_t)M * D * 4);
    float* nx2f = (float*)alloc((size_t)M * D * 4);
    float* y    = (float*)alloc((size_t)M * D * 4);
    float* outF = (float*)d_out;

    auto cvt = [&](const float* src, u16* dst, int n) {
        cvt_bf16_kernel<<<(n + 255) / 256, 256, 0, stream>>>(src, dst, n);
    };
    cvt(wq, wq_b, D * D);   cvt(wk, wk_b, D * D);
    cvt(wv, wv_b, D * D);   cvt(wo, wo_b, D * D);
    cvt(w1, w1_b, D * DFF); cvt(w2, w2_b, DFF * D);
    cvt(w3, w3_b, D * D);

    // LN1
    ln_kernel<<<M, 256, 0, stream>>>(x, g1, be1, nx_b, nullptr);

    // Q, K(transposed layout), V projections
    dim3 gDD(D / 64, M / 64);
    gemm_wmma<EPI_BF16>  <<<gDD, 128, 0, stream>>>(nx_b, wq_b, bq, nullptr, nullptr, nullptr, Q_b,  M, D, D);
    gemm_wmma<EPI_KTRANS><<<gDD, 128, 0, stream>>>(nx_b, wk_b, bk, nullptr, nullptr, nullptr, Kt_b, M, D, D);
    gemm_wmma<EPI_BF16>  <<<gDD, 128, 0, stream>>>(nx_b, wv_b, bv, nullptr, nullptr, nullptr, V_b,  M, D, D);

    // Attention
    attn_kernel<<<Bb * H * (S / 16), 256, 0, stream>>>(Q_b, Kt_b, V_b, rel, ctx_b);

    // O-projection + residual:  xa = x + ctx @ wo + bo
    gemm_wmma<EPI_F32_RES><<<gDD, 128, 0, stream>>>(ctx_b, wo_b, bo, x, nullptr, xa, nullptr, M, D, D);

    // LN2 (need bf16 + fp32 copies)
    ln_kernel<<<M, 256, 0, stream>>>(xa, g2, be2, nx2_b, nx2f);

    // FFN: h1 = gelu(nx2 @ w1 + b1)
    dim3 gDF(DFF / 64, M / 64);
    gemm_wmma<EPI_GELU><<<gDF, 128, 0, stream>>>(nx2_b, w1_b, b1, nullptr, nullptr, nullptr, h1_b, M, DFF, D);

    // y = nx2 + h1 @ w2 + b2
    gemm_wmma<EPI_F32_RES><<<gDD, 128, 0, stream>>>(h1_b, w2_b, b2, nx2f, nullptr, y, nullptr, M, D, DFF);

    // LNf
    ln_kernel<<<M, 256, 0, stream>>>(y, gf, bfb, yn_b, nullptr);

    // out = xa + y + (ln(y) @ w3 + b3)
    gemm_wmma<EPI_F32_RES2><<<gDD, 128, 0, stream>>>(yn_b, w3_b, b3, xa, y, outF, nullptr, M, D, D);

    (void)in_sizes; (void)n_in; (void)out_size; (void)ws_size;
}
